// SE3Transformer_30580167147937
// MI455X (gfx1250) — matrix operand order repaired and checked by reading, not compiled, not run on hardware
//
#include <hip/hip_runtime.h>
#include <hip/hip_bf16.h>

// ---------------------------------------------------------------------------
// SE(3)-transformer forward for MI455X (gfx1250, wave32, WMMA + TDM).
// B=2, N=512, D=256, H=8, DH=32, L=4, DFF=1024, ML=512, NS=1000.
// GEMM-shaped math -> v_wmma_f32_16x16x32_f16 (fp16 in, fp32 acc).
// A-tiles staged to LDS via tensor_load_to_lds (TDM) when available.
// Bandwidth rewrites: pos_w via (i-j) table, sq via h.h^T GEMM.
// ---------------------------------------------------------------------------

#define Bc 2
#define Nc 512
#define Dc 256
#define Hc 8
#define DHc 32
#define Lc 4
#define DFFc 1024
#define MLc 512
#define BNc (Bc * Nc)

typedef __attribute__((ext_vector_type(16))) _Float16 v16h;
typedef __attribute__((ext_vector_type(8)))  float    v8f;
typedef __attribute__((ext_vector_type(4)))  unsigned u32x4;
typedef __attribute__((ext_vector_type(8)))  int      i32x8;
typedef __attribute__((ext_vector_type(4)))  int      i32x4;

#if defined(__has_builtin)
#  if __has_builtin(__builtin_amdgcn_tensor_load_to_lds) && \
      __has_builtin(__builtin_amdgcn_s_wait_tensorcnt)
#    define SE3_HAVE_TDM 1
#  endif
#endif
#ifndef SE3_HAVE_TDM
#  define SE3_HAVE_TDM 0
#endif

#define WMMA_F16(A, B, C) \
  __builtin_amdgcn_wmma_f32_16x16x32_f16(false, (A), false, (B), (short)0, (C), false, false)

// Load one 16x32 fp16 A/B fragment from a row-major matrix tile.
// ISA 7.12.2: lane L -> row (or col) L%16; halves e map to
// K = base + (e<8 ? e : e+8), base = (L<16) ? 0 : 8.
__device__ inline v16h load_frag16(const _Float16* __restrict__ base, int ld, int lane) {
  int r  = lane & 15;
  int kb = (lane & 16) ? 8 : 0;
  const _Float16* p = base + (long)r * ld + kb;
  v16h f;
#pragma unroll
  for (int i = 0; i < 8; ++i) f[i] = p[i];          // K = base+0..7  (16B contiguous)
#pragma unroll
  for (int i = 0; i < 8; ++i) f[8 + i] = p[16 + i]; // K = base+16..23
  return f;
}

#if SE3_HAVE_TDM
// Issue a TDM load of `nelem` contiguous fp16 elements from `gsrc` into LDS at
// `ldsoff` (1-D descriptor per CDNA5 ISA Sec 8.3/8.4).
__device__ inline void se3_tdm_load_1d(unsigned long long gsrc, unsigned ldsoff,
                                       unsigned nelem) {
  u32x4 g0;
  g0[0] = 1u;                                        // count=1 (valid), no gather
  g0[1] = ldsoff;                                    // LDS byte address
  g0[2] = (unsigned)(gsrc & 0xFFFFFFFFu);            // global_addr[31:0]
  g0[3] = (unsigned)((gsrc >> 32) & 0x1FFFFFFu)      // global_addr[56:32]
        | 0x80000000u;                               // type=2 ("image")
  i32x8 g1;
  g1[0] = (int)(1u << 16);                           // data_size=1 (2 bytes)
  g1[1] = (int)((nelem & 0xFFFFu) << 16);            // tensor_dim0 lo16
  g1[2] = (int)((nelem >> 16) | (1u << 16));         // tensor_dim0 hi16 | tensor_dim1=1
  g1[3] = (int)((nelem & 0xFFFFu) << 16);            // tile_dim0 = nelem (<= 16384)
  g1[4] = 0;                                         // tile_dim1=0, tile_dim2=0 (1-D)
  g1[5] = (int)nelem;                                // tensor_dim0_stride lo32
  g1[6] = 0;
  g1[7] = 0;
  i32x4 gz = {0, 0, 0, 0};
#if defined(__clang_major__) && (__clang_major__ >= 23)
  i32x8 gz8 = {0, 0, 0, 0, 0, 0, 0, 0};
  __builtin_amdgcn_tensor_load_to_lds(g0, g1, gz, gz, gz8, 0);
#else
  __builtin_amdgcn_tensor_load_to_lds(g0, g1, gz, gz, 0);
#endif
}
#endif

// ---------------------------------------------------------------------------
// fp32 -> fp16 conversion (weights, once per launch)
// ---------------------------------------------------------------------------
__global__ void se3_f2h(const float* __restrict__ s, _Float16* __restrict__ d, int n) {
  int i = blockIdx.x * blockDim.x + threadIdx.x;
  int stride = gridDim.x * blockDim.x;
  for (; i < n; i += stride) d[i] = (_Float16)s[i];
}

// ---------------------------------------------------------------------------
// te_proj[b,dout] = sum_d relu(tf_b*W_time[d]+b_time[d]) * W_in[dout,3+d]
// ---------------------------------------------------------------------------
__global__ void se3_teproj(const int* __restrict__ t, const float* __restrict__ Wt,
                           const float* __restrict__ bt, const float* __restrict__ Win,
                           float* __restrict__ teproj) {
  int idx = blockIdx.x * blockDim.x + threadIdx.x;
  if (idx >= Bc * Dc) return;
  int b = idx / Dc, dout = idx % Dc;
  float tf = (float)t[b] / 1000.0f;
  float acc = 0.f;
  for (int d = 0; d < Dc; ++d) {
    float te = fmaxf(tf * Wt[d] + bt[d], 0.f);
    acc += te * Win[dout * (3 + Dc) + 3 + d];
  }
  teproj[idx] = acc;
}

// h0 = relu(x @ W_in[:, :3]^T + te_proj + b_in)
__global__ void se3_h0(const float* __restrict__ x, const float* __restrict__ Win,
                       const float* __restrict__ bin, const float* __restrict__ teproj,
                       float* __restrict__ h32, _Float16* __restrict__ h16) {
  int row = blockIdx.x;       // BN
  int d   = threadIdx.x;      // D
  int b   = row >> 9;         // /N
  const float* xr = x + row * 3;
  const float* wr = Win + d * (3 + Dc);
  float v = teproj[b * Dc + d] + bin[d] + xr[0] * wr[0] + xr[1] * wr[1] + xr[2] * wr[2];
  v = fmaxf(v, 0.f);
  h32[(long)row * Dc + d] = v;
  h16[(long)row * Dc + d] = (_Float16)v;
}

// ---------------------------------------------------------------------------
// rproj[h, p] = rel_pos[p] . pos_W[l,h] + pos_b[l,h],  p in [0,1023)
// (pos_w[h,i,j] = rproj[h, i-j+511] -- avoids 268MB pe materialization)
// ---------------------------------------------------------------------------
__global__ void se3_rproj(const float* __restrict__ relpos, const float* __restrict__ posW,
                          const float* __restrict__ posB, float* __restrict__ rproj) {
  int idx = blockIdx.x * blockDim.x + threadIdx.x;
  if (idx >= (2 * MLc - 1) * Hc) return;
  int p = idx / Hc, hh = idx % Hc;
  float acc = posB[hh];
  for (int d = 0; d < Dc; ++d) acc += relpos[(long)p * Dc + d] * posW[hh * Dc + d];
  rproj[hh * (2 * MLc - 1) + p] = acc;
}

// nrm[row] = |h_row|^2
__global__ void se3_rownorm(const float* __restrict__ h, float* __restrict__ nrm) {
  int row = blockIdx.x * blockDim.x + threadIdx.x;
  if (row >= BNc) return;
  float s = 0.f;
  for (int d = 0; d < Dc; ++d) { float v = h[(long)row * Dc + d]; s += v * v; }
  nrm[row] = s;
}

// ---------------------------------------------------------------------------
// Generic batched WMMA GEMM: C[M,N] = A(M,K) * W(N,K)^T (+bias)(+relu)
// A,W fp16 row-major. One wave per 16x64 C tile. grid = (M/16, N/64, batches)
// A tile (contiguous 16*K halves) staged to LDS via TDM; W streamed from HBM.
// ---------------------------------------------------------------------------
__device__ inline void se3_store_tile(v8f acc, int bz, int m0, int n0t, int n, int mo,
                                      const float* __restrict__ bias,
                                      float* __restrict__ Cf, _Float16* __restrict__ Ch,
                                      int Nout, long sC, int relu) {
#pragma unroll
  for (int r = 0; r < 8; ++r) {
    int mm = m0 + mo + r, nn = n0t + n;
    float v = acc[r];
    if (bias) v += bias[nn];
    if (relu) v = fmaxf(v, 0.f);
    long off = (long)bz * sC + (long)mm * Nout + nn;
    if (Cf) Cf[off] = v;
    if (Ch) Ch[off] = (_Float16)v;
  }
}

__global__ void __launch_bounds__(32)
se3_gemm16(const _Float16* __restrict__ A, const _Float16* __restrict__ W,
           const float* __restrict__ bias, float* __restrict__ Cf,
           _Float16* __restrict__ Ch, int K, int Nout,
           long sA, long sW, long sC, int relu) {
  __shared__ __align__(128) _Float16 sAtile[16 * 1024];   // up to K=1024 (32 KB)
  int lane = threadIdx.x;
  int m0 = blockIdx.x * 16;
  int n0 = blockIdx.y * 64;
  int bz = blockIdx.z;
  const _Float16* Ab = A + (long)bz * sA + (long)m0 * K;
  const _Float16* Wb = W + (long)bz * sW;

  // Stage contiguous 16xK A tile into LDS.
#if SE3_HAVE_TDM
  se3_tdm_load_1d((unsigned long long)Ab,
                  (unsigned)(unsigned long long)(const void*)sAtile,
                  (unsigned)(16 * K));
  __builtin_amdgcn_s_wait_tensorcnt(0);
#else
  {
    const unsigned* src = (const unsigned*)Ab;
    unsigned* dst = (unsigned*)sAtile;
    for (int i = lane; i < 8 * K; i += 32) dst[i] = src[i];
    __syncthreads();
  }
#endif

  v8f a0 = {}, a1 = {}, a2 = {}, a3 = {};
  for (int k = 0; k < K; k += 32) {
    if (k + 32 < K) {
      // prefetch next k-chunk of the 4 W row-blocks (global_prefetch_b8)
      int pr = lane & 15;
#pragma unroll
      for (int tt = 0; tt < 4; ++tt)
        __builtin_prefetch(Wb + (long)(n0 + 16 * tt + pr) * K + k + 32, 0, 3);
    }
    v16h af = load_frag16(sAtile + k, K, lane);
    v16h b0 = load_frag16(Wb + (long)(n0 +  0) * K + k, K, lane);
    v16h b1 = load_frag16(Wb + (long)(n0 + 16) * K + k, K, lane);
    v16h b2 = load_frag16(Wb + (long)(n0 + 32) * K + k, K, lane);
    v16h b3 = load_frag16(Wb + (long)(n0 + 48) * K + k, K, lane);
    a0 = WMMA_F16(af, b0, a0);
    a1 = WMMA_F16(af, b1, a1);
    a2 = WMMA_F16(af, b2, a2);
    a3 = WMMA_F16(af, b3, a3);
  }
  int n  = lane & 15;
  int mo = (lane >> 4) << 3;
  se3_store_tile(a0, bz, m0, n0 +  0, n, mo, bias, Cf, Ch, Nout, sC, relu);
  se3_store_tile(a1, bz, m0, n0 + 16, n, mo, bias, Cf, Ch, Nout, sC, relu);
  se3_store_tile(a2, bz, m0, n0 + 32, n, mo, bias, Cf, Ch, Nout, sC, relu);
  se3_store_tile(a3, bz, m0, n0 + 48, n, mo, bias, Cf, Ch, Nout, sC, relu);
}

// ---------------------------------------------------------------------------
// Attention: one wave per (b, h, 16-row i-tile).
// logits = q.k^T/sqrt(DH) + geo(sq) + pos(i-j) (+mask), softmax, ao = P.V
// qkv layout: (B, N, 768) with [q|k|v] each (H,DH) packed.
// ---------------------------------------------------------------------------
__global__ void __launch_bounds__(32)
se3_attn(const _Float16* __restrict__ qkv, const float* __restrict__ G,
         const float* __restrict__ nrm, const float* __restrict__ rproj,
         const float* __restrict__ geoW, const float* __restrict__ geoB,
         const unsigned char* __restrict__ mask, _Float16* __restrict__ ao) {
  __shared__ float    slog[16 * Nc];      // 32 KB logits strip
  __shared__ _Float16 sp[16 * Nc];        // 16 KB fp16 probabilities
  __shared__ float    sred[64];

  int lane = threadIdx.x;
  int i0 = blockIdx.x * 16;
  int hh = blockIdx.y;
  int b  = blockIdx.z;

  const _Float16* qb = qkv + ((long)(b * Nc + i0)) * (3 * Dc) + hh * DHc;
  v16h qf = load_frag16(qb, 3 * Dc, lane);   // 16 x 32 (K = DH)

  int n  = lane & 15;
  int mo = (lane >> 4) << 3;
  float gw0 = geoW[hh * 2 + 0], gw1 = geoW[hh * 2 + 1], gb = geoB[hh];
  const float scale = 0.17677669529663687f; // 1/sqrt(32)

  for (int j0 = 0; j0 < Nc; j0 += 16) {
    v16h kf = load_frag16(qkv + ((long)(b * Nc + j0)) * (3 * Dc) + Dc + hh * DHc,
                          3 * Dc, lane);
    v8f acc = {};
    acc = WMMA_F16(qf, kf, acc);
    int j = j0 + n;
    float nj = nrm[b * Nc + j];
    bool mj = mask[b * Nc + j] != 0;
#pragma unroll
    for (int r = 0; r < 8; ++r) {
      int i = i0 + mo + r;
      float sq = nrm[b * Nc + i] + nj - 2.0f * G[((long)(b * Nc + i)) * Nc + j];
      float nv = (sq > 0.f) ? sqrtf(sq) : 0.f;
      float lg = acc[r] * scale + gw0 * nv + gw1 * sq + gb
               + rproj[hh * (2 * MLc - 1) + (i - j + MLc - 1)];
      slog[(mo + r) * Nc + j] = mj ? lg : -__builtin_inff();
    }
  }
  __syncthreads();

  // softmax: row m handled by lane pair (m, m+16), each over half the columns
  int m = lane & 15;
  int c0 = (lane >> 4) * (Nc / 2);
  float mx = -__builtin_inff();
  for (int j = 0; j < Nc / 2; ++j) mx = fmaxf(mx, slog[m * Nc + c0 + j]);
  sred[lane] = mx;
  __syncthreads();
  mx = fmaxf(sred[m], sred[16 + m]);
  float sum = 0.f;
  for (int j = 0; j < Nc / 2; ++j) {
    float e = __expf(slog[m * Nc + c0 + j] - mx);
    slog[m * Nc + c0 + j] = e;
    sum += e;
  }
  sred[32 + lane] = sum;
  __syncthreads();
  float inv = 1.0f / (sred[32 + m] + sred[48 + m]);
  for (int j = 0; j < Nc / 2; ++j)
    sp[m * Nc + c0 + j] = (_Float16)(slog[m * Nc + c0 + j] * inv);
  __syncthreads();

  // ao = P (16xN) x V (NxDH) via WMMA over K
  v8f o0 = {}, o1 = {};
  for (int kb = 0; kb < Nc; kb += 32) {
    v16h pf = load_frag16(sp + kb, Nc, lane);
    const _Float16* vb = qkv + ((long)(b * Nc + kb)) * (3 * Dc) + 2 * Dc + hh * DHc;
    v16h v0, v1;
    int vn = lane & 15, vk = (lane & 16) ? 8 : 0;
#pragma unroll
    for (int i = 0; i < 8; ++i) {
      v0[i]     = vb[(long)(vk + i) * (3 * Dc) + vn];
      v0[8 + i] = vb[(long)(vk + 16 + i) * (3 * Dc) + vn];
      v1[i]     = vb[(long)(vk + i) * (3 * Dc) + 16 + vn];
      v1[8 + i] = vb[(long)(vk + 16 + i) * (3 * Dc) + 16 + vn];
    }
    o0 = WMMA_F16(pf, v0, o0);
    o1 = WMMA_F16(pf, v1, o1);
  }
#pragma unroll
  for (int r = 0; r < 8; ++r) {
    int i = i0 + mo + r;
    ao[((long)(b * Nc + i)) * Dc + hh * DHc + n]      = (_Float16)o0[r];
    ao[((long)(b * Nc + i)) * Dc + hh * DHc + 16 + n] = (_Float16)o1[r];
  }
}

// ---------------------------------------------------------------------------
// h = LayerNorm(h + (mask?)*add) ; also emit fp16 copy. One row per block.
// ---------------------------------------------------------------------------
__global__ void se3_ln(float* __restrict__ h, _Float16* __restrict__ h16,
                       const float* __restrict__ add, const unsigned char* __restrict__ mask,
                       const float* __restrict__ g, const float* __restrict__ bta) {
  __shared__ float s1[Dc], s2[Dc];
  int row = blockIdx.x, d = threadIdx.x;
  float a = add[(long)row * Dc + d];
  if (mask) a *= (mask[row] ? 1.f : 0.f);
  float v = h[(long)row * Dc + d] + a;
  s1[d] = v; s2[d] = v * v;
  __syncthreads();
  for (int s = Dc / 2; s > 0; s >>= 1) {
    if (d < s) { s1[d] += s1[d + s]; s2[d] += s2[d + s]; }
    __syncthreads();
  }
  float mean = s1[0] / (float)Dc;
  float var  = s2[0] / (float)Dc - mean * mean;
  float y = (v - mean) * rsqrtf(var + 1e-5f) * g[d] + bta[d];
  h[(long)row * Dc + d]   = y;
  h16[(long)row * Dc + d] = (_Float16)y;
}

// out = (h @ W_out^T + b_out) * mask
__global__ void se3_out(const float* __restrict__ h, const float* __restrict__ Wo,
                        const float* __restrict__ bo, const unsigned char* __restrict__ mask,
                        float* __restrict__ out) {
  int row = blockIdx.x * blockDim.x + threadIdx.x;
  if (row >= BNc) return;
  float mv = mask[row] ? 1.f : 0.f;
  for (int c = 0; c < 3; ++c) {
    float acc = bo[c];
    for (int d = 0; d < Dc; ++d) acc += h[(long)row * Dc + d] * Wo[c * Dc + d];
    out[row * 3 + c] = acc * mv;
  }
}

// ---------------------------------------------------------------------------
// Host orchestration
// ---------------------------------------------------------------------------
extern "C" void kernel_launch(void* const* d_in, const int* in_sizes, int n_in,
                              void* d_out, int out_size, void* d_ws, size_t ws_size,
                              hipStream_t stream) {
  (void)in_sizes; (void)n_in; (void)out_size; (void)ws_size;

  const float* x       = (const float*)d_in[0];
  const int*   t       = (const int*)d_in[1];
  const unsigned char* mask = (const unsigned char*)d_in[2];
  const float* W_time  = (const float*)d_in[3];
  const float* b_time  = (const float*)d_in[4];
  const float* W_in    = (const float*)d_in[5];
  const float* b_in    = (const float*)d_in[6];
  const float* rel_pos = (const float*)d_in[7];
  const float* qkv_W   = (const float*)d_in[8];
  const float* qkv_b   = (const float*)d_in[9];
  const float* ao_W    = (const float*)d_in[10];
  const float* ao_b    = (const float*)d_in[11];
  const float* geo_W   = (const float*)d_in[12];
  const float* geo_b   = (const float*)d_in[13];
  const float* pos_W   = (const float*)d_in[14];
  const float* pos_b   = (const float*)d_in[15];
  const float* ln1_g   = (const float*)d_in[16];
  const float* ln1_b   = (const float*)d_in[17];
  const float* ln2_g   = (const float*)d_in[18];
  const float* ln2_b   = (const float*)d_in[19];
  const float* fc1_W   = (const float*)d_in[20];
  const float* fc1_b   = (const float*)d_in[21];
  const float* fc2_W   = (const float*)d_in[22];
  const float* fc2_b   = (const float*)d_in[23];
  const float* W_out   = (const float*)d_in[24];
  const float* b_out   = (const float*)d_in[25];
  float* out = (float*)d_out;

  // workspace carve-up
  char* ws = (char*)d_ws;
  size_t off = 0;
  auto take = [&](size_t bytes) { size_t o = off; off += (bytes + 255) & ~(size_t)255; return o; };
  float*    h32     = (float*)   (ws + take((size_t)BNc * Dc * 4));
  _Float16* h16     = (_Float16*)(ws + take((size_t)BNc * Dc * 2));
  float*    teproj  = (float*)   (ws + take((size_t)Bc * Dc * 4));
  _Float16* qkvW16  = (_Float16*)(ws + take((size_t)Lc * 3 * Dc * Dc * 2));
  _Float16* aoW16   = (_Float16*)(ws + take((size_t)Lc * Dc * Dc * 2));
  _Float16* fc1W16  = (_Float16*)(ws + take((size_t)Lc * DFFc * Dc * 2));
  _Float16* fc2W16  = (_Float16*)(ws + take((size_t)Lc * Dc * DFFc * 2));
  _Float16* qkv16   = (_Float16*)(ws + take((size_t)BNc * 3 * Dc * 2));
  float*    Gmat    = (float*)   (ws + take((size_t)Bc * Nc * Nc * 4));
  float*    nrm     = (float*)   (ws + take((size_t)BNc * 4));
  float*    rproj   = (float*)   (ws + take((size_t)Hc * (2 * MLc - 1) * 4));
  _Float16* ao16    = (_Float16*)(ws + take((size_t)BNc * Dc * 2));
  float*    tmp32   = (float*)   (ws + take((size_t)BNc * Dc * 4));   // aoproj / ffn2
  _Float16* ffn116  = (_Float16*)(ws + take((size_t)BNc * DFFc * 2));

  // 1) fp16 weight conversion (deterministic, per launch)
  {
    int n;
    n = Lc * 3 * Dc * Dc; se3_f2h<<<(n + 255) / 256, 256, 0, stream>>>(qkv_W, qkvW16, n);
    n = Lc * Dc * Dc;     se3_f2h<<<(n + 255) / 256, 256, 0, stream>>>(ao_W,  aoW16,  n);
    n = Lc * DFFc * Dc;   se3_f2h<<<(n + 255) / 256, 256, 0, stream>>>(fc1_W, fc1W16, n);
    n = Lc * Dc * DFFc;   se3_f2h<<<(n + 255) / 256, 256, 0, stream>>>(fc2_W, fc2W16, n);
  }

  // 2) input embedding
  se3_teproj<<<(Bc * Dc + 255) / 256, 256, 0, stream>>>(t, W_time, b_time, W_in, teproj);
  se3_h0<<<BNc, Dc, 0, stream>>>(x, W_in, b_in, teproj, h32, h16);

  // 3) layers
  for (int l = 0; l < Lc; ++l) {
    const _Float16* qW16 = qkvW16 + (size_t)l * 3 * Dc * Dc;
    const _Float16* aW16 = aoW16  + (size_t)l * Dc * Dc;
    const _Float16* f1W  = fc1W16 + (size_t)l * DFFc * Dc;
    const _Float16* f2W  = fc2W16 + (size_t)l * Dc * DFFc;

    // relative-position projection table (H x 1023)
    se3_rproj<<<((2 * MLc - 1) * Hc + 255) / 256, 256, 0, stream>>>(
        rel_pos, pos_W + (size_t)l * Hc * Dc, pos_b + (size_t)l * Hc, rproj);

    // row norms + Gram matrix G = h.h^T (batched WMMA GEMM)
    se3_rownorm<<<(BNc + 255) / 256, 256, 0, stream>>>(h32, nrm);
    se3_gemm16<<<dim3(Nc / 16, Nc / 64, Bc), 32, 0, stream>>>(
        h16, h16, nullptr, Gmat, nullptr, Dc, Nc,
        (long)Nc * Dc, (long)Nc * Dc, (long)Nc * Nc, 0);

    // qkv = h @ qkv_W^T + b  (fp16 out for attention)
    se3_gemm16<<<dim3(BNc / 16, (3 * Dc) / 64, 1), 32, 0, stream>>>(
        h16, qW16, qkv_b + (size_t)l * 3 * Dc, nullptr, qkv16,
        Dc, 3 * Dc, 0, 0, 0, 0);

    // attention (WMMA q.k^T + softmax + WMMA P.V), per (b,h,i-tile) wave
    se3_attn<<<dim3(Nc / 16, Hc, Bc), 32, 0, stream>>>(
        qkv16, Gmat, nrm, rproj,
        geo_W + (size_t)l * Hc * 2, geo_b + (size_t)l * Hc, mask, ao16);

    // ao projection + residual LN1
    se3_gemm16<<<dim3(BNc / 16, Dc / 64, 1), 32, 0, stream>>>(
        ao16, aW16, ao_b + (size_t)l * Dc, tmp32, nullptr, Dc, Dc, 0, 0, 0, 0);
    se3_ln<<<BNc, Dc, 0, stream>>>(h32, h16, tmp32, nullptr,
                                   ln1_g + (size_t)l * Dc, ln1_b + (size_t)l * Dc);

    // FFN: relu(h W1^T + b1) W2^T + b2, masked, residual LN2
    se3_gemm16<<<dim3(BNc / 16, DFFc / 64, 1), 32, 0, stream>>>(
        h16, f1W, fc1_b + (size_t)l * DFFc, nullptr, ffn116, Dc, DFFc, 0, 0, 0, 1);
    se3_gemm16<<<dim3(BNc / 16, Dc / 64, 1), 32, 0, stream>>>(
        ffn116, f2W, fc2_b + (size_t)l * Dc, tmp32, nullptr, DFFc, Dc, 0, 0, 0, 0);
    se3_ln<<<BNc, Dc, 0, stream>>>(h32, h16, tmp32, mask,
                                   ln2_g + (size_t)l * Dc, ln2_b + (size_t)l * Dc);
  }

  // 4) output head
  se3_out<<<(BNc + 255) / 256, 256, 0, stream>>>(h32, W_out, b_out, mask, out);
}
